// VQCodebook_5068061409454
// MI455X (gfx1250) — matrix operand order repaired and checked by reading, not compiled, not run on hardware
//
#include <hip/hip_runtime.h>
#include <hip/hip_bf16.h>

// ---------------------------------------------------------------------------
// VQ codebook lookup on gfx1250 (MI455X).
//   z:         (B=32, D=256, T=2048) fp32
//   embedding: (K=1024, D=256) fp32
// Distance argmin via v_wmma_f32_16x16x32_bf16 (||e||^2 - 2 z.e ; ||z||^2
// cancels in argmin). Codebook chunks staged into LDS by the Tensor Data
// Mover (tensor_load_to_lds + s_wait_tensorcnt) when the builtin exists,
// with LDS row padding done by the TDM pad engine. Gather/loss/EMA in fp32.
// ---------------------------------------------------------------------------

typedef __attribute__((ext_vector_type(16))) __bf16 v16bf;
typedef __attribute__((ext_vector_type(8)))  __bf16 v8bf;
typedef __attribute__((ext_vector_type(8)))  float  v8f;
typedef __attribute__((ext_vector_type(4)))  unsigned int u32x4;
typedef __attribute__((ext_vector_type(8)))  int i32x8;
typedef __attribute__((ext_vector_type(4)))  int i32x4;

constexpr int Bv = 32, Dv = 256, Tv = 2048, Kv = 1024;
constexpr int Nrows = Bv * Tv;                 // 65536
constexpr int ROWS_PER_WG = 128;               // t-block per workgroup
constexpr int ECHUNK = 128;                    // codes staged in LDS per pass
constexpr int LSTRIDE = 264;                   // padded LDS row stride (bf16 elems)
                                               // = 512B data + 16B pad per row

// output float offsets (tuple concatenated flat, return order)
constexpr size_t ZQ_OFF   = 0;
constexpr size_t ZQ_N     = (size_t)Bv * Dv * Tv;          // 16777216
constexpr size_t TOK_OFF  = ZQ_OFF + ZQ_N;                 // 16777216
constexpr size_t LOSS_OFF = TOK_OFF + (size_t)Bv * Tv;     // 16842752
constexpr size_t EMB_OFF  = LOSS_OFF + 1;                  // 16842753
constexpr size_t CNT_OFF  = EMB_OFF + (size_t)Kv * Dv;     // 17104897
constexpr size_t WGT_OFF  = CNT_OFF + Kv;                  // 17105921

// workspace float offsets
constexpr size_t WS_ENORM = 0;                   // 1024 f
constexpr size_t WS_CNT   = 1024;                // 1024 f
constexpr size_t WS_SUMS  = 2048;                // 262144 f
constexpr size_t WS_LOSS  = 264192;              // 1 f
constexpr size_t WS_INV   = 264193;              // 1024 f
constexpr size_t WS_EMBB  = 265220;              // bf16 region (16B aligned)
constexpr size_t WS_ZERO_BASE = WS_CNT;          // zero counts+sums+loss per launch
constexpr size_t WS_ZERO_N    = 1024 + (size_t)Kv * Dv + 1;   // 263169

#if __has_builtin(__builtin_amdgcn_tensor_load_to_lds) && \
    __has_builtin(__builtin_amdgcn_s_wait_tensorcnt)
#define VQ_USE_TDM 1
#else
#define VQ_USE_TDM 0
#endif

__global__ void vq_zero_kernel(float* p, int n) {
    int i = blockIdx.x * blockDim.x + threadIdx.x;
    if (i < n) p[i] = 0.0f;
}

// one block per code: ||e_k||^2 and bf16 codebook copy
__global__ void vq_prep_kernel(const float* __restrict__ emb,
                               float* __restrict__ enorm,
                               __bf16* __restrict__ embb) {
    __shared__ float red[256];
    int k = blockIdx.x, d = threadIdx.x;
    float e = emb[k * Dv + d];
    embb[k * Dv + d] = (__bf16)e;
    red[d] = e * e;
    __syncthreads();
    for (int s = 128; s > 0; s >>= 1) {
        if (d < s) red[d] += red[d + s];
        __syncthreads();
    }
    if (d == 0) enorm[k] = red[0];
}

#if VQ_USE_TDM
// One TDM descriptor: 2D tile of 128 rows x 256 bf16 (512B) from the bf16
// codebook into LDS, padding LDS by 16B after every 512B row so the LDS
// row stride is LSTRIDE(264) bf16. D# layout per CDNA5 ISA ch.8.
__device__ __forceinline__ void vq_tdm_load_chunk(const __bf16* gsrc,
                                                  unsigned lds_byte_off) {
    unsigned long long ga = (unsigned long long)(uintptr_t)gsrc;
    u32x4 g0 = {
        1u,                                          // count=1 (valid), load, user
        lds_byte_off,                                // lds_addr [63:32]
        (unsigned)(ga & 0xffffffffu),                // global_addr [95:64]
        (unsigned)((ga >> 32) & 0x01ffffffu) | (2u << 30) // addr[56:32] | type=2
    };
    i32x8 g1 = {
        // data_size=1 (2B) | pad_enable | pad_interval=6 (128 DW = 512B)
        // | pad_amount=3 (4 DW = 16B); workgroup_mask=0 (not in cluster)
        (int)((1u << 16) | (1u << 20) | (6u << 22) | (3u << 25)),
        (int)(256u << 16),                           // tensor_dim0 = 256 ([79:48])
        (int)(1024u << 16),                          // tensor_dim1 = 1024 ([111:80])
        (int)(256u << 16),                           // tile_dim0 = 256 ([127:112])
        128,                                         // tile_dim1 = 128 ([143:128])
        256,                                         // tensor_dim0_stride = 256
        0, 0
    };
    i32x4 gz = { 0, 0, 0, 0 };
#if __clang_major__ >= 23
    i32x8 gz8 = { 0, 0, 0, 0, 0, 0, 0, 0 };
    __builtin_amdgcn_tensor_load_to_lds(g0, g1, gz, gz, gz8, 0);
#else
    __builtin_amdgcn_tensor_load_to_lds(g0, g1, gz, gz, 0);
#endif
}
#endif

__launch_bounds__(256)
__global__ void vq_main_kernel(const float* __restrict__ z,
                               const float* __restrict__ emb,
                               const float* __restrict__ enorm,
                               const __bf16* __restrict__ embb,
                               float* __restrict__ out,
                               float* __restrict__ ws_counts,
                               float* __restrict__ ws_sums,
                               float* __restrict__ ws_loss) {
    __shared__ __align__(16) __bf16 lds_z[ROWS_PER_WG * LSTRIDE];
    __shared__ __align__(16) __bf16 lds_e[ECHUNK * LSTRIDE];
    __shared__ int   lds_tok[ROWS_PER_WG];
    __shared__ float lds_red[256];

    const int tid  = threadIdx.x;
    const int lane = tid & 31;
    const int wave = tid >> 5;         // 8 waves, 16 rows each
    const int tblk = blockIdx.x;       // 0..15
    const int b    = blockIdx.y;       // 0..31
    const int t0   = tblk * ROWS_PER_WG;

    // ---- stage z tile (transpose (d,t) -> [t_local][d]) as bf16 ----------
    // consecutive tid -> consecutive t (coalesced global reads)
    {
        const float* zb = z + (size_t)b * Dv * Tv + t0;
        for (int it = 0; it < (Dv * ROWS_PER_WG) / 256; ++it) {
            int idx = it * 256 + tid;            // over (d, t_local)
            int tl  = idx & (ROWS_PER_WG - 1);
            int d   = idx >> 7;
            float v = zb[(size_t)d * Tv + tl];
            lds_z[tl * LSTRIDE + d] = (__bf16)v;
        }
    }
    __syncthreads();

    // ---- load this wave's A tile (16 rows x 256 K) into registers --------
    // A layout (16-bit, 16x32): lane m=lane&15, half=lane>>4,
    //   elems 0..7  -> K = k0 + half*8 + 0..7
    //   elems 8..15 -> K = k0 + 16 + half*8 + 0..7
    v16bf a_reg[8];
    {
        const int am = lane & 15, ah = lane >> 4;
        const int rowbase = (wave * 16 + am) * LSTRIDE;
#pragma unroll
        for (int kc = 0; kc < 8; ++kc) {
            int base = rowbase + kc * 32 + ah * 8;
            v8bf lo = *(const v8bf*)&lds_z[base];
            v8bf hi = *(const v8bf*)&lds_z[base + 16];
            a_reg[kc] = __builtin_shufflevector(lo, hi,
                0, 1, 2, 3, 4, 5, 6, 7, 8, 9, 10, 11, 12, 13, 14, 15);
        }
    }

    float minv[8];
    int   mini[8];
#pragma unroll
    for (int v = 0; v < 8; ++v) { minv[v] = 3.4e38f; mini[v] = 0; }

    const int m    = lane & 15;
    const int half = lane >> 4;
#if VQ_USE_TDM
    const unsigned lds_e_off = (unsigned)(unsigned long long)(uintptr_t)(void*)lds_e;
#endif

    // ---- sweep codebook in LDS-staged chunks of 128 codes ----------------
    for (int c0 = 0; c0 < Kv; c0 += ECHUNK) {
        __syncthreads();   // previous chunk fully consumed
#if VQ_USE_TDM
        // Tensor Data Mover: one DMA descriptor stages the whole chunk,
        // padding the LDS rows in-flight. Issued by wave 0 only (EXEC is
        // ignored by tensor ops; TENSORcnt is per-wave).
        if (wave == 0) {
            vq_tdm_load_chunk(embb + (size_t)c0 * Dv, lds_e_off);
            __builtin_amdgcn_s_wait_tensorcnt(0);
        }
#else
        // fallback: manual 16B vector copies through VGPRs
        {
            const uint4* src = (const uint4*)(embb + (size_t)c0 * Dv);
            for (int it = 0; it < (ECHUNK * Dv * 2 / 16) / 256; ++it) { // 16 iters
                int c   = it * 256 + tid;        // 16B chunk id
                int row = c >> 5;                // 32 chunks per 512B row
                int off = (c & 31) * 8;          // bf16 elem offset in row
                uint4 v = src[(size_t)row * (Dv / 8) + (c & 31)];
                *(uint4*)&lds_e[row * LSTRIDE + off] = v;
            }
        }
#endif
        if (c0 + ECHUNK < Kv)
            __builtin_prefetch(embb + (size_t)(c0 + ECHUNK) * Dv, 0, 0);
        __syncthreads();

        // 8 column tiles of 16 codes each
        for (int ct = 0; ct < 8; ++ct) {
            v8f acc = {};
            const int colbase = (ct * 16 + m) * LSTRIDE;
#pragma unroll
            for (int kc = 0; kc < 8; ++kc) {
                // B layout (16-bit, 32x16): lane col = lane&15,
                //   elems 0..7 -> K = k0 + half*8, elems 8..15 -> K = k0 + 16 + half*8
                int base = colbase + kc * 32 + half * 8;
                v8bf lo = *(const v8bf*)&lds_e[base];
                v8bf hi = *(const v8bf*)&lds_e[base + 16];
                v16bf bt = __builtin_shufflevector(lo, hi,
                    0, 1, 2, 3, 4, 5, 6, 7, 8, 9, 10, 11, 12, 13, 14, 15);
                acc = __builtin_amdgcn_wmma_f32_16x16x32_bf16(
                    false, a_reg[kc], false, bt, (short)0, acc, false, false);
            }
            const int col = c0 + ct * 16 + m;    // this lane's code index
            const float en = enorm[col];
#pragma unroll
            for (int v = 0; v < 8; ++v) {
                float s = en - 2.0f * acc[v];    // ||e||^2 - 2 z.e
                if (s < minv[v]) { minv[v] = s; mini[v] = col; }
            }
        }
    }

    // ---- cross-lane argmin within each 16-lane half (cols live on lanes) -
#pragma unroll
    for (int v = 0; v < 8; ++v) {
        float val = minv[v];
        int   idx = mini[v];
        for (int off = 8; off > 0; off >>= 1) {
            float ov = __shfl_xor(val, off, 16);
            int   oi = __shfl_xor(idx, off, 16);
            if (ov < val || (ov == val && oi < idx)) { val = ov; idx = oi; }
        }
        if (m == 0) {
            int row_local = wave * 16 + v + 8 * half;   // 0..127
            lds_tok[row_local] = idx;
            int t = t0 + row_local;
            out[TOK_OFF + (size_t)b * Tv + t] = (float)idx;
            atomicAdd(&ws_counts[idx], 1.0f);
        }
    }
    __syncthreads();

    // ---- gather z_q (fp32), loss partials, EMA segment sums --------------
    float lpart = 0.0f;
    {
        const float* zb  = z + (size_t)b * Dv * Tv + t0;
        float*       zqb = out + ZQ_OFF + (size_t)b * Dv * Tv + t0;
        for (int it = 0; it < (Dv * ROWS_PER_WG) / 256; ++it) {
            int idx = it * 256 + tid;
            int tl  = idx & (ROWS_PER_WG - 1);
            int d   = idx >> 7;
            int tok = lds_tok[tl];
            float e  = emb[(size_t)tok * Dv + d];
            float zv = zb[(size_t)d * Tv + tl];
            zqb[(size_t)d * Tv + tl] = e;        // z_q (STE == gathered e)
            float df = zv - e;
            lpart += df * df;
            atomicAdd(&ws_sums[(size_t)tok * Dv + d], zv);
        }
    }
    lds_red[tid] = lpart;
    __syncthreads();
    for (int s = 128; s > 0; s >>= 1) {
        if (tid < s) lds_red[tid] += lds_red[tid + s];
        __syncthreads();
    }
    if (tid == 0) atomicAdd(ws_loss, lds_red[0]);
}

// single block, 1024 threads: EMA counts, smoothing, loss scalar
__global__ void vq_finalize_a(const float* __restrict__ ema_count,
                              const float* __restrict__ ws_counts,
                              const float* __restrict__ ws_loss,
                              float* __restrict__ out,
                              float* __restrict__ ws_inv) {
    __shared__ float red[1024];
    int k = threadIdx.x;
    float cnt = 0.99f * ema_count[k] + 0.01f * ws_counts[k];
    out[CNT_OFF + k] = cnt;
    red[k] = cnt;
    __syncthreads();
    for (int s = 512; s > 0; s >>= 1) {
        if (k < s) red[k] += red[k + s];
        __syncthreads();
    }
    float n = red[0];
    float smoothed = (cnt + 1e-5f) / (n + Kv * 1e-5f) * n;
    ws_inv[k] = 1.0f / smoothed;
    if (k == 0)
        out[LOSS_OFF] = 1.25f * ws_loss[0] / (float)((size_t)Nrows * Dv);
}

__global__ void vq_finalize_b(const float* __restrict__ ema_weight,
                              const float* __restrict__ ws_sums,
                              const float* __restrict__ ws_inv,
                              float* __restrict__ out) {
    int i = blockIdx.x * blockDim.x + threadIdx.x;   // over K*D
    int k = i >> 8;
    float w = 0.99f * ema_weight[i] + 0.01f * ws_sums[i];
    out[WGT_OFF + i] = w;
    out[EMB_OFF + i] = w * ws_inv[k];
}

extern "C" void kernel_launch(void* const* d_in, const int* in_sizes, int n_in,
                              void* d_out, int out_size, void* d_ws, size_t ws_size,
                              hipStream_t stream) {
    const float* z          = (const float*)d_in[0];
    const float* embedding  = (const float*)d_in[1];
    const float* ema_count  = (const float*)d_in[2];
    const float* ema_weight = (const float*)d_in[3];
    float* out = (float*)d_out;

    float*  ws_f     = (float*)d_ws;
    float*  ws_enorm = ws_f + WS_ENORM;
    float*  ws_cnt   = ws_f + WS_CNT;
    float*  ws_sums  = ws_f + WS_SUMS;
    float*  ws_loss  = ws_f + WS_LOSS;
    float*  ws_inv   = ws_f + WS_INV;
    __bf16* ws_embb  = (__bf16*)(ws_f + WS_EMBB);

    // zero accumulators (counts + sums + loss)
    vq_zero_kernel<<<(int)((WS_ZERO_N + 255) / 256), 256, 0, stream>>>(
        ws_f + WS_ZERO_BASE, (int)WS_ZERO_N);

    // codebook norms + bf16 copy
    vq_prep_kernel<<<Kv, 256, 0, stream>>>(embedding, ws_enorm, ws_embb);

    // main WMMA distance/argmin + gather/loss/segment sums
    vq_main_kernel<<<dim3(Tv / ROWS_PER_WG, Bv), 256, 0, stream>>>(
        z, embedding, ws_enorm, ws_embb, out, ws_cnt, ws_sums, ws_loss);

    // EMA finalize
    vq_finalize_a<<<1, 1024, 0, stream>>>(ema_count, ws_cnt, ws_loss, out, ws_inv);
    vq_finalize_b<<<(Kv * Dv) / 256, 256, 0, stream>>>(ema_weight, ws_sums, ws_inv, out);
}
